// RegionProposalNetwork_77506979823627
// MI455X (gfx1250) — matrix operand order repaired and checked by reading, not compiled, not run on hardware
//
#include <hip/hip_runtime.h>
#include <hip/hip_bf16.h>

typedef __attribute__((ext_vector_type(16))) _Float16 v16h;
typedef __attribute__((ext_vector_type(8)))  _Float16 v8h;
typedef __attribute__((ext_vector_type(8)))  float    v8f;

#define HMAP 64
#define WMAP 64
#define CIN 256
#define HID 512
#define KTOT 2304       // 9 * 256
#define NA 42
#define MANC (64*64*NA) // 172032 anchors per image
#define BIMG 8
#define NGT 20
#define LDA 40          // 32 + 8 halfs row stride (80B, multiple of 16B)

// ---------- helpers ----------

__device__ __forceinline__ v16h frag16(const _Float16* p0, const _Float16* p1) {
    v8h a = *(const v8h*)p0;
    v8h b = *(const v8h*)p1;
    return __builtin_shufflevector(a, b, 0,1,2,3,4,5,6,7,8,9,10,11,12,13,14,15);
}

__device__ __forceinline__ float softplus_f(float x) {
    return fmaxf(x, 0.f) + log1pf(expf(-fabsf(x)));
}

__device__ __forceinline__ float hash01(unsigned v) {
    v = (v ^ 61u) ^ (v >> 16); v *= 9u; v ^= v >> 4;
    v *= 0x27d4eb2du; v ^= v >> 15;
    return (float)v * 2.3283064365386963e-10f;
}

__device__ __forceinline__ void anchor_box(int m, float* a) {
    const float SC[6] = {2.f, 2.5f, 3.f, 3.5f, 4.f, 5.f};
    const float RT[7] = {0.5f, 1.5f, 2.f, 2.5f, 3.f, 3.5f, 4.f};
    int aidx = m % NA;
    int cell = m / NA;
    int x = cell & 63, y = cell >> 6;
    float s = SC[aidx / 7], r = RT[aidx % 7];
    float h = 16.f * s * sqrtf(r);
    float w = 16.f * s * sqrtf(1.f / r);
    float cy = 8.f + (float)y * 16.f;
    float cx = 8.f + (float)x * 16.f;
    a[0] = cy - 0.5f * h; a[1] = cx - 0.5f * w;
    a[2] = cy + 0.5f * h; a[3] = cx + 0.5f * w;
}

__device__ __forceinline__ float iou_fn(const float* a, const float* g) {
    float tly = fmaxf(a[0], g[0]), tlx = fmaxf(a[1], g[1]);
    float bry = fminf(a[2], g[2]), brx = fminf(a[3], g[3]);
    float wy = fmaxf(bry - tly, 0.f), wx = fmaxf(brx - tlx, 0.f);
    float inter = wy * wx;
    float areaA = (a[2] - a[0]) * (a[3] - a[1]);
    float areaG = (g[2] - g[0]) * (g[3] - g[1]);
    return inter / (areaA + areaG - inter);
}

// project gt to anchor space, keep -1 padding, g_safe substitution
__device__ __forceinline__ bool load_gt(const float* gt, int b, int n, float* g) {
    const float* p = gt + ((size_t)b * NGT + n) * 4;
    float g0 = (p[0] == -1.f) ? -1.f : p[0] * (1.f / 16.f);
    float g1 = (p[1] == -1.f) ? -1.f : p[1] * (1.f / 16.f);
    float g2 = (p[2] == -1.f) ? -1.f : p[2] * (1.f / 16.f);
    float g3 = (p[3] == -1.f) ? -1.f : p[3] * (1.f / 16.f);
    bool valid = (g0 >= 0.f);
    if (valid) { g[0] = g0; g[1] = g1; g[2] = g2; g[3] = g3; }
    else       { g[0] = 0.f; g[1] = 0.f; g[2] = 1.f; g[3] = 1.f; }
    return valid;
}

// ---------- kernel 0: zero scalar accumulators ----------
__global__ void zero_scal_kernel(float* scal) {
    if (threadIdx.x < 8) scal[threadIdx.x] = 0.f;
}

// ---------- kernel 1: 3x3 conv via implicit-GEMM WMMA, fused bias+ReLU, f16 out ----------
// grid: (256 pixel tiles, 8 M tiles), block 256 (8 waves)
__global__ __launch_bounds__(256) void conv3x3_wmma_kernel(
    const float* __restrict__ fmap, const float* __restrict__ w1,
    const float* __restrict__ b1, _Float16* __restrict__ xh)
{
    __shared__ __align__(16) _Float16 As[64 * LDA];   // [m][k]
    __shared__ __align__(16) _Float16 Bs[128 * LDA];  // [n][k]

    const int tid   = threadIdx.x;
    const int lane  = tid & 31;
    const int wave  = tid >> 5;
    const int waveM = wave >> 2;       // 0..1
    const int waveN = wave & 3;        // 0..3
    const int mBase = blockIdx.y * 64; // output channel base
    const int ptile = blockIdx.x;      // 0..255
    const int bimg  = ptile >> 5;
    const int y0    = (ptile & 31) * 2;

    v8f acc[2][2] = {};

    const int am  = tid >> 2;          // 0..63 : weight row
    const int akq = (tid & 3) * 8;     // 0,8,16,24
    const int bk  = tid >> 3;          // 0..31 : K row of patch tile
    const int bx0 = (tid & 7) * 16;    // chunk of 16 pixels
    const int ry  = bx0 >> 6;          // row within 2-row tile
    const int x0  = bx0 & 63;

    for (int s = 0; s < KTOT / 32; ++s) {
        const int kBase = s * 32;
        const int t     = kBase >> 8;      // tap 0..8 (constant within step)
        const int cBase = kBase & 255;
        const int ky = t / 3 - 1, kx = t % 3 - 1;

        // ---- stage A: weights 64x32 ----
        {
            const int o = mBase + am;
            const float* wp = w1 + (size_t)o * KTOT + (size_t)(cBase + akq) * 9 + t;
            v8h av;
            #pragma unroll
            for (int i = 0; i < 8; ++i) av[i] = (_Float16)wp[i * 9];
            *(v8h*)&As[am * LDA + akq] = av;
            if (s + 1 < KTOT / 32) {
                const int t2 = ((s + 1) * 32) >> 8;
                const int c2 = ((s + 1) * 32) & 255;
                __builtin_prefetch(w1 + (size_t)o * KTOT + (size_t)(c2 + akq) * 9 + t2, 0, 1);
            }
        }
        // ---- stage B: input patches 32x128, transposed into [n][k] ----
        {
            const int c  = cBase + bk;
            const int yy = y0 + ry + ky;
            const bool yok = (yy >= 0) && (yy < HMAP);
            const float* src = fmap + (((size_t)bimg * CIN + c) * HMAP + (yok ? yy : 0)) * WMAP;
            #pragma unroll
            for (int i = 0; i < 16; ++i) {
                const int xx = x0 + i + kx;
                float v = 0.f;
                if (yok && xx >= 0 && xx < WMAP) v = src[xx];
                Bs[(bx0 + i) * LDA + bk] = (_Float16)v;
            }
        }
        __syncthreads();

        // ---- fragments + WMMA ----
        const int hi = lane >> 4;
        const int ml = lane & 15;
        v16h afr[2], bfr[2];
        #pragma unroll
        for (int ti = 0; ti < 2; ++ti) {
            const int m = waveM * 32 + ti * 16 + ml;
            afr[ti] = frag16(&As[m * LDA + hi * 8], &As[m * LDA + 16 + hi * 8]);
        }
        #pragma unroll
        for (int tj = 0; tj < 2; ++tj) {
            const int n = waveN * 32 + tj * 16 + ml;
            bfr[tj] = frag16(&Bs[n * LDA + hi * 8], &Bs[n * LDA + 16 + hi * 8]);
        }
        #pragma unroll
        for (int ti = 0; ti < 2; ++ti)
            #pragma unroll
            for (int tj = 0; tj < 2; ++tj)
                acc[ti][tj] = __builtin_amdgcn_wmma_f32_16x16x32_f16(
                    false, afr[ti], false, bfr[tj], (short)0, acc[ti][tj], false, false);
        __syncthreads();
    }

    // ---- epilogue: bias + ReLU, store f16 activations ----
    const int hi = lane >> 4;
    const int ml = lane & 15;
    #pragma unroll
    for (int ti = 0; ti < 2; ++ti) {
        #pragma unroll
        for (int tj = 0; tj < 2; ++tj) {
            const int n = waveN * 32 + tj * 16 + ml;
            const int p = (ptile & 31) * 128 + n;   // pixel index within image
            #pragma unroll
            for (int r = 0; r < 8; ++r) {
                const int o = mBase + waveM * 32 + ti * 16 + r + 8 * hi;
                float v = acc[ti][tj][r] + b1[o];
                v = fmaxf(v, 0.f);
                xh[((size_t)bimg * HID + o) * 4096 + p] = (_Float16)v;
            }
        }
    }
}

// ---------- kernel 2: fused 1x1 conf+reg heads via WMMA ----------
// grid: (256 pixel tiles, 14 M tiles of 16), block 128 (4 waves)
__global__ __launch_bounds__(128) void heads_wmma_kernel(
    const _Float16* __restrict__ xh,
    const float* __restrict__ conf_w, const float* __restrict__ conf_b,
    const float* __restrict__ reg_w,  const float* __restrict__ reg_b,
    float* __restrict__ conf_flat, float* __restrict__ reg_flat)
{
    __shared__ __align__(16) _Float16 As[16 * LDA];
    __shared__ __align__(16) _Float16 Bs[128 * LDA];

    const int tid  = threadIdx.x;
    const int lane = tid & 31;
    const int wave = tid >> 5;         // 0..3
    const int mtile = blockIdx.y;      // 0..13
    const int ptile = blockIdx.x;      // 0..255
    const int bimg  = ptile >> 5;
    const int pBase = (ptile & 31) * 128;

    v8f acc[2] = {};

    const int ar  = tid >> 3;          // 0..15
    const int akq = (tid & 7) * 4;     // 0..28
    const int bkk = tid >> 2;          // 0..31
    const int bn0 = (tid & 3) * 32;    // 0,32,64,96

    for (int s = 0; s < HID / 32; ++s) {
        const int kB = s * 32;
        // stage combined head weights (rows >= 210 are zero pad)
        {
            const int R = mtile * 16 + ar;
            #pragma unroll
            for (int i = 0; i < 4; ++i) {
                const int c = kB + akq + i;
                float v = 0.f;
                if (R < NA)        v = conf_w[(size_t)R * HID + c];
                else if (R < 210)  v = reg_w[(size_t)(R - NA) * HID + c];
                As[ar * LDA + akq + i] = (_Float16)v;
            }
        }
        // stage activations, transposed to [n][k]
        {
            const _Float16* src = xh + ((size_t)bimg * HID + (kB + bkk)) * 4096 + pBase + bn0;
            #pragma unroll
            for (int i = 0; i < 32; ++i) Bs[(bn0 + i) * LDA + bkk] = src[i];
        }
        __syncthreads();

        const int hi = lane >> 4;
        const int ml = lane & 15;
        v16h af = frag16(&As[ml * LDA + hi * 8], &As[ml * LDA + 16 + hi * 8]);
        #pragma unroll
        for (int tj = 0; tj < 2; ++tj) {
            const int n = wave * 32 + tj * 16 + ml;
            v16h bf = frag16(&Bs[n * LDA + hi * 8], &Bs[n * LDA + 16 + hi * 8]);
            acc[tj] = __builtin_amdgcn_wmma_f32_16x16x32_f16(
                false, af, false, bf, (short)0, acc[tj], false, false);
        }
        __syncthreads();
    }

    const int hi = lane >> 4;
    const int ml = lane & 15;
    #pragma unroll
    for (int tj = 0; tj < 2; ++tj) {
        const int n = wave * 32 + tj * 16 + ml;
        const size_t prow = (size_t)bimg * 4096 + pBase + n;
        #pragma unroll
        for (int r = 0; r < 8; ++r) {
            const int R = mtile * 16 + r + 8 * hi;
            float v = acc[tj][r];
            if (R < NA)       conf_flat[prow * NA + R] = v + conf_b[R];
            else if (R < 210) reg_flat[prow * 168 + (R - NA)] = v + reg_b[R - NA];
        }
    }
}

// ---------- kernel 3: per-GT max IoU over all anchors ----------
__global__ __launch_bounds__(256) void gtmax_kernel(const float* __restrict__ gt,
                                                    float* __restrict__ mpg)
{
    const int b = blockIdx.x / NGT, n = blockIdx.x % NGT;
    float g[4];
    const bool valid = load_gt(gt, b, n, g);
    float best = 0.f;
    if (valid) {
        for (int m = threadIdx.x; m < MANC; m += 256) {
            float a[4]; anchor_box(m, a);
            best = fmaxf(best, iou_fn(a, g));
        }
    }
    __shared__ float red[256];
    red[threadIdx.x] = best; __syncthreads();
    for (int o = 128; o > 0; o >>= 1) {
        if (threadIdx.x < o) red[threadIdx.x] = fmaxf(red[threadIdx.x], red[threadIdx.x + o]);
        __syncthreads();
    }
    if (threadIdx.x == 0) mpg[blockIdx.x] = red[0];
}

// ---------- kernel 4: anchor assignment + positive losses ----------
// scal: [0]=P [1]=negCount [2]=Spos [3]=Sreg [4]=Sneg [5]=selCnt
__global__ __launch_bounds__(256) void assign_kernel(
    const float* __restrict__ gt, const float* __restrict__ mpg,
    const float* __restrict__ conf_flat, const float* __restrict__ reg_flat,
    unsigned char* __restrict__ negf, float* __restrict__ scal)
{
    const int idx = blockIdx.x * 256 + threadIdx.x;   // b*MANC + m
    float lP = 0.f, lN = 0.f, lSp = 0.f, lSr = 0.f;
    if (idx < BIMG * MANC) {
        const int b = idx / MANC;
        const int m = idx % MANC;
        float a[4]; anchor_box(m, a);
        float best = -1.f;
        float bg[4] = {0.f, 0.f, 1.f, 1.f};
        bool pos = false;
        for (int n = 0; n < NGT; ++n) {
            float g[4];
            const bool valid = load_gt(gt, b, n, g);
            const float iou = valid ? iou_fn(a, g) : 0.f;
            const float mg = mpg[b * NGT + n];
            if ((iou == mg && mg > 0.f) || iou > 0.5f) pos = true;
            if (iou > best) { best = iou; bg[0] = g[0]; bg[1] = g[1]; bg[2] = g[2]; bg[3] = g[3]; }
        }
        const bool neg = best < 0.3f;
        negf[idx] = neg ? 1 : 0;
        lN = neg ? 1.f : 0.f;
        if (pos) {
            lP = 1.f;
            lSp = softplus_f(-conf_flat[idx]);
            const float acy = 0.5f * (a[0] + a[2]), acx = 0.5f * (a[1] + a[3]);
            const float ah = a[2] - a[0], aw = a[3] - a[1];
            const float gcy = 0.5f * (bg[0] + bg[2]), gcx = 0.5f * (bg[1] + bg[3]);
            const float gh = bg[2] - bg[0], gw = bg[3] - bg[1];
            float off[4];
            off[0] = (gcy - acy) / ah;
            off[1] = (gcx - acx) / aw;
            off[2] = logf(gh / ah);
            off[3] = logf(gw / aw);
            #pragma unroll
            for (int j = 0; j < 4; ++j) {
                const float d = reg_flat[(size_t)idx * 4 + j] - off[j];
                const float ad = fabsf(d);
                lSr += (ad < 1.f) ? 0.5f * ad * ad : ad - 0.5f;
            }
        }
    }
    __shared__ float red[256];
    float vals[4] = {lP, lN, lSp, lSr};
    #pragma unroll
    for (int q = 0; q < 4; ++q) {
        red[threadIdx.x] = vals[q]; __syncthreads();
        for (int o = 128; o > 0; o >>= 1) {
            if (threadIdx.x < o) red[threadIdx.x] += red[threadIdx.x + o];
            __syncthreads();
        }
        if (threadIdx.x == 0 && red[0] != 0.f) atomicAdd(&scal[q], red[0]);
        __syncthreads();
    }
}

// ---------- kernel 5: sampled negative cls loss ----------
__global__ __launch_bounds__(256) void negsel_kernel(
    const unsigned char* __restrict__ negf, const float* __restrict__ conf_flat,
    float* __restrict__ scal)
{
    const int idx = blockIdx.x * 256 + threadIdx.x;
    const float P = scal[0];
    const float NT = scal[1];
    float lS = 0.f, lC = 0.f;
    if (idx < BIMG * MANC && negf[idx]) {
        if (hash01((unsigned)idx) * NT < P) {
            lC = 1.f;
            lS = softplus_f(conf_flat[idx]);
        }
    }
    __shared__ float red[256];
    float vals[2] = {lS, lC};
    #pragma unroll
    for (int q = 0; q < 2; ++q) {
        red[threadIdx.x] = vals[q]; __syncthreads();
        for (int o = 128; o > 0; o >>= 1) {
            if (threadIdx.x < o) red[threadIdx.x] += red[threadIdx.x + o];
            __syncthreads();
        }
        if (threadIdx.x == 0 && red[0] != 0.f) atomicAdd(&scal[4 + q], red[0]);
        __syncthreads();
    }
}

// ---------- kernel 6: finalize scalar loss ----------
__global__ void finalize_kernel(const float* __restrict__ scal, float* __restrict__ out) {
    if (threadIdx.x == 0) {
        const float P = scal[0];
        const float cls = 0.5f * (scal[2] / P + scal[4] / scal[5]);
        const float reg = scal[3] / (P * 4.f);
        out[0] = cls + reg;
    }
}

// ---------- host launcher ----------
extern "C" void kernel_launch(void* const* d_in, const int* in_sizes, int n_in,
                              void* d_out, int out_size, void* d_ws, size_t ws_size,
                              hipStream_t stream) {
    const float* fmap = (const float*)d_in[0];
    const float* gt   = (const float*)d_in[1];
    /* d_in[2] = gt_classes (unused by the loss) */
    const float* w1 = (const float*)d_in[3];
    const float* b1 = (const float*)d_in[4];
    const float* cw = (const float*)d_in[5];
    const float* cb = (const float*)d_in[6];
    const float* rw = (const float*)d_in[7];
    const float* rb = (const float*)d_in[8];
    float* out = (float*)d_out;

    char* ws = (char*)d_ws;
    size_t off = 0;
    _Float16* xh = (_Float16*)(ws + off); off += (size_t)BIMG * HID * 4096 * sizeof(_Float16);
    float* conf_flat = (float*)(ws + off); off += (size_t)BIMG * MANC * sizeof(float);
    float* reg_flat  = (float*)(ws + off); off += (size_t)BIMG * MANC * 4 * sizeof(float);
    float* mpg  = (float*)(ws + off); off += 256 * sizeof(float);
    float* scal = (float*)(ws + off); off += 64 * sizeof(float);
    unsigned char* negf = (unsigned char*)(ws + off); off += (size_t)BIMG * MANC;

    zero_scal_kernel<<<1, 64, 0, stream>>>(scal);
    conv3x3_wmma_kernel<<<dim3(256, 8), 256, 0, stream>>>(fmap, w1, b1, xh);
    heads_wmma_kernel<<<dim3(256, 14), 128, 0, stream>>>(xh, cw, cb, rw, rb, conf_flat, reg_flat);
    gtmax_kernel<<<BIMG * NGT, 256, 0, stream>>>(gt, mpg);
    const int totalAnc = BIMG * MANC;
    assign_kernel<<<(totalAnc + 255) / 256, 256, 0, stream>>>(gt, mpg, conf_flat, reg_flat, negf, scal);
    negsel_kernel<<<(totalAnc + 255) / 256, 256, 0, stream>>>(negf, conf_flat, scal);
    finalize_kernel<<<1, 1, 0, stream>>>(scal, out);
}
